// AdvancedCommunication_2345052144034
// MI455X (gfx1250) — matrix operand order repaired and checked by reading, not compiled, not run on hardware
//
#include <hip/hip_runtime.h>
#include <math.h>

#define HW 65536
#define IMG_W 256
#define NAG 5
#define CEPS 1e-8f

typedef __attribute__((ext_vector_type(16))) __bf16 v16bf;
typedef __attribute__((ext_vector_type(8)))  float  v8f;

__device__ __forceinline__ v8f vsplat8(float s) {
  v8f v;
#pragma unroll
  for (int i = 0; i < 8; ++i) v[i] = s;
  return v;
}

__device__ __forceinline__ v8f wmma_bf16(v16bf a, v16bf b, v8f c) {
  // 8 args: (neg_a, A, neg_b, B, c_mod, C, reuse_a, reuse_b) -> v_wmma_f32_16x16x32_bf16
  return __builtin_amdgcn_wmma_f32_16x16x32_bf16(false, a, false, b, (short)0, c,
                                                 false, false);
}

// A fragment from row-major [M][K] LDS slab; rowptr = &A[laneM*stride + c0].
// Contiguous K order used here (same as frag_b): lanes0-15 hold K = c0+0..15; lanes16-31 hold K = c0+16..31.
__device__ __forceinline__ v16bf frag_a(const __bf16* rowptr, int grp) {
  v16bf a;
  const __bf16* p = rowptr + 16 * grp;
#pragma unroll
  for (int e = 0; e < 16; ++e) a[e] = p[e];
  return a;
}

// B fragment from row-major [N][K] LDS slab; rowptr = &B[laneN*stride + c0].
// Contiguous K order used here (same as frag_a): lanes0-15 hold K = c0+0..15; lanes16-31 hold K = c0+16..31.
__device__ __forceinline__ v16bf frag_b(const __bf16* rowptr, int grp) {
  v16bf b;
  const __bf16* p = rowptr + 16 * grp;
#pragma unroll
  for (int e = 0; e < 16; ++e) b[e] = p[e];
  return b;
}

// ---------------------------------------------------------------------------
// 1) Contribution: padded cosine similarities (exact f32, memory-bound)
// ---------------------------------------------------------------------------
__global__ void __launch_bounds__(256) k_contribution(
    const float* __restrict__ ev, const float* __restrict__ ef,
    const float* __restrict__ ed, const float* __restrict__ cv,
    const float* __restrict__ cf, const float* __restrict__ cd,
    float* __restrict__ out) {
  int idx = blockIdx.x * blockDim.x + threadIdx.x;
  if (idx >= NAG * HW) return;
  int n = idx / HW, p = idx - n * HW;
  float dv = 0.f, nv = 0.f, df = 0.f, nf = 0.f, dd = 0.f, nd = 0.f, ref = 0.f;
#pragma unroll 4
  for (int c = 0; c < 8; ++c) {
    float e = ev[c * HW + p], x = cv[(n * 8 + c) * HW + p];
    dv += x * e; nv += x * x; ref += e * e;
  }
#pragma unroll 8
  for (int c = 0; c < 64; ++c) {
    float e = ef[c * HW + p], x = cf[(n * 64 + c) * HW + p];
    df += x * e; nf += x * x; ref += e * e;
  }
#pragma unroll 4
  for (int c = 0; c < 8; ++c) {
    float e = ed[c * HW + p], x = cd[(n * 8 + c) * HW + p];
    dd += x * e; nd += x * x; ref += e * e;
  }
  float dr = fmaxf(sqrtf(ref), CEPS);
  out[(n * 3 + 0) * HW + p] = (dv / (fmaxf(sqrtf(nv), CEPS) * dr) + 1.f) * 0.5f;
  out[(n * 3 + 1) * HW + p] = (df / (fmaxf(sqrtf(nf), CEPS) * dr) + 1.f) * 0.5f;
  out[(n * 3 + 2) * HW + p] = (dd / (fmaxf(sqrtf(nd), CEPS) * dr) + 1.f) * 0.5f;
}

// ---------------------------------------------------------------------------
// 2) Fused Q/K 1x1 convs + score softmax. One wave per 16-pixel tile.
// ---------------------------------------------------------------------------
__global__ void __launch_bounds__(32) k_qk(
    const float* __restrict__ ef, const float* __restrict__ cv,
    const float* __restrict__ cf, const float* __restrict__ cd,
    const float* __restrict__ qw, const float* __restrict__ kw1,
    const float* __restrict__ kw2, const float* __restrict__ kw3,
    float* __restrict__ out) {
  __shared__ __bf16 sAf[16 * 64];  // ego_feat tile  [px][c]   (for q)
  __shared__ __bf16 sAc[16 * 64];  // collab_feat    [px][c]   (for k2)
  __shared__ __bf16 sAv[16 * 32];  // collab_vox     [px][cpad32]
  __shared__ __bf16 sAd[16 * 32];  // collab_det     [px][cpad32]
  __shared__ __bf16 sBq[16 * 64];  // q_w   [o][c]
  __shared__ __bf16 sB2[16 * 64];  // k_w2  [o][c]
  __shared__ __bf16 sB1[16 * 32];  // k_w1  [o][cpad32]
  __shared__ __bf16 sB3[16 * 32];  // k_w3  [o][cpad32]

  int bid = blockIdx.x;
  int x0 = (bid & 15) << 4;
  int y = (bid >> 4) & 255;
  int n = bid >> 12;
  int p0 = y * IMG_W + x0;
  int lane = threadIdx.x;
  int laneM = lane & 15, grp = lane >> 4;

  for (int i = lane; i < 16 * 64; i += 32) {
    int c = i >> 4, px = i & 15;
    sAf[px * 64 + c] = (__bf16)ef[c * HW + p0 + px];
    sAc[px * 64 + c] = (__bf16)cf[(n * 64 + c) * HW + p0 + px];
    sBq[i] = (__bf16)qw[i];   // same [o][c] layout as global
    sB2[i] = (__bf16)kw2[i];
  }
  for (int i = lane; i < 16 * 32; i += 32) {
    int c = i >> 4, px = i & 15;
    float vv = (c < 8) ? cv[(n * 8 + c) * HW + p0 + px] : 0.f;
    float vd = (c < 8) ? cd[(n * 8 + c) * HW + p0 + px] : 0.f;
    sAv[px * 32 + c] = (__bf16)vv;
    sAd[px * 32 + c] = (__bf16)vd;
    int o = i >> 5, cc = i & 31;
    sB1[i] = (cc < 8) ? (__bf16)kw1[o * 8 + cc] : (__bf16)0.f;
    sB3[i] = (cc < 8) ? (__bf16)kw3[o * 8 + cc] : (__bf16)0.f;
  }
  __syncthreads();

  v8f q = vsplat8(0.f), s1 = vsplat8(0.f), s2 = vsplat8(0.f), s3 = vsplat8(0.f);
#pragma unroll
  for (int ks = 0; ks < 2; ++ks) {
    int c0 = ks * 32;
    v16bf a = frag_a(sAf + laneM * 64 + c0, grp);
    v16bf b = frag_b(sBq + laneM * 64 + c0, grp);
    q = wmma_bf16(a, b, q);
    a = frag_a(sAc + laneM * 64 + c0, grp);
    b = frag_b(sB2 + laneM * 64 + c0, grp);
    s2 = wmma_bf16(a, b, s2);
  }
  {
    v16bf a = frag_a(sAv + laneM * 32, grp);
    v16bf b = frag_b(sB1 + laneM * 32, grp);
    s1 = wmma_bf16(a, b, s1);
    a = frag_a(sAd + laneM * 32, grp);
    b = frag_b(sB3 + laneM * 32, grp);
    s3 = wmma_bf16(a, b, s3);
  }

  // scores[m] = sum_o q[m][o]*k[m][o]: reduce over the 16-lane N dimension
  v8f d1, d2, d3;
#pragma unroll
  for (int i = 0; i < 8; ++i) {
    d1[i] = q[i] * s1[i]; d2[i] = q[i] * s2[i]; d3[i] = q[i] * s3[i];
  }
#pragma unroll
  for (int m = 1; m <= 8; m <<= 1) {
#pragma unroll
    for (int i = 0; i < 8; ++i) {
      d1[i] += __shfl_xor(d1[i], m, 32);
      d2[i] += __shfl_xor(d2[i], m, 32);
      d3[i] += __shfl_xor(d3[i], m, 32);
    }
  }
  const float scale = 0.25f;  // 1/sqrt(16)
#pragma unroll
  for (int i = 0; i < 8; ++i) {
    float a1 = d1[i] * scale, a2 = d2[i] * scale, a3 = d3[i] * scale;
    float mx = fmaxf(a1, fmaxf(a2, a3));
    float e1 = expf(a1 - mx), e2 = expf(a2 - mx), e3 = expf(a3 - mx);
    float inv = 1.f / (e1 + e2 + e3);
    if (laneM == i) {  // lanes i and 16+i cover m = i and m = 8+i
      int p = p0 + grp * 8 + i;
      out[(n * 3 + 0) * HW + p] = e1 * inv;
      out[(n * 3 + 1) * HW + p] = e2 * inv;
      out[(n * 3 + 2) * HW + p] = e3 * inv;
    }
  }
}

// ---------------------------------------------------------------------------
// 3) Spatial branch: conv3x3(80->64)+ReLU, 1x1(64->1)+sigmoid (implicit GEMM)
//    Block = 128 threads (4 waves), M-tile = 32 pixels, Kpad = 96 per tap.
// ---------------------------------------------------------------------------
__global__ void __launch_bounds__(128) k_spatial(
    const float* __restrict__ cv, const float* __restrict__ cf,
    const float* __restrict__ cd, const float* __restrict__ w1,
    const float* __restrict__ b1, const float* __restrict__ w2,
    const float* __restrict__ b2, float* __restrict__ out) {
  __shared__ __bf16 As[102 * 96];  // [r*34+xx][cpad96] input halo patch
  __shared__ __bf16 Bs[64 * 96];   // [o][cpad96] per-tap weight slab
  __shared__ float  Cis[32 * 64];  // [px][o] relu intermediate

  int bid = blockIdx.x;
  int x0 = (bid & 7) << 5;
  int y = (bid >> 3) & 255;
  int n = bid >> 11;
  int tid = threadIdx.x;
  int lane = tid & 31, wave = tid >> 5;
  int laneM = lane & 15, grp = lane >> 4;

  // stage input patch (coalesced along x within each channel run)
  for (int i = tid; i < 96 * 102; i += 128) {
    int c = i / 102, row = i - c * 102;
    int r = row / 34, xx = row - r * 34;
    int gy = y + r - 1, gx = x0 + xx - 1;
    float v = 0.f;
    if (c < 80 && (unsigned)gy < 256u && (unsigned)gx < 256u) {
      const float* src;
      if (c < 8)       src = cv + (n * 8 + c) * HW;
      else if (c < 72) src = cf + (n * 64 + (c - 8)) * HW;
      else             src = cd + (n * 8 + (c - 72)) * HW;
      v = src[gy * IMG_W + gx];
    }
    As[row * 96 + c] = (__bf16)v;
  }

  int o0 = wave * 16 + laneM;  // this lane's output channel (N)
  v8f acc0 = vsplat8(b1[o0]);  // M-tile 0 (px 0..15)
  v8f acc1 = acc0;             // M-tile 1 (px 16..31)

  for (int tap = 0; tap < 9; ++tap) {
    int ky = tap / 3, kx = tap - ky * 3;
    __syncthreads();
    for (int i = tid; i < 64 * 96; i += 128) {
      int o = i / 96, c = i - o * 96;
      Bs[i] = (c < 80) ? (__bf16)w1[((o * 80 + c) * 3 + ky) * 3 + kx]
                       : (__bf16)0.f;
    }
    __syncthreads();
#pragma unroll
    for (int ks = 0; ks < 3; ++ks) {
      int c0 = ks * 32;
      const __bf16* ar = As + (ky * 34 + laneM + kx) * 96 + c0;
      v16bf a0 = frag_a(ar, grp);
      v16bf a1 = frag_a(ar + 16 * 96, grp);
      v16bf b  = frag_b(Bs + o0 * 96 + c0, grp);
      acc0 = wmma_bf16(a0, b, acc0);
      acc1 = wmma_bf16(a1, b, acc1);
    }
  }
  __syncthreads();
#pragma unroll
  for (int i = 0; i < 8; ++i) {
    int m0 = grp * 8 + i;
    Cis[m0 * 64 + o0] = fmaxf(acc0[i], 0.f);
    Cis[(16 + m0) * 64 + o0] = fmaxf(acc1[i], 0.f);
  }
  __syncthreads();
  if (tid < 32) {
    float s = b2[0];
#pragma unroll 8
    for (int o = 0; o < 64; ++o) s += Cis[tid * 64 + o] * w2[o];
    out[n * HW + y * IMG_W + x0 + tid] = 1.f / (1.f + expf(-s));
  }
}

// ---------------------------------------------------------------------------
// 4) Semantic branch: conv3x3(64->128)+ReLU then 1x1(128->32), both WMMA.
// ---------------------------------------------------------------------------
__global__ void __launch_bounds__(128) k_semantic(
    const float* __restrict__ cf, const float* __restrict__ w1,
    const float* __restrict__ b1, const float* __restrict__ w2,
    const float* __restrict__ b2, float* __restrict__ out) {
  __shared__ __bf16 As[102 * 64];   // [r*34+xx][c]
  __shared__ __bf16 Bs[128 * 64];   // [o][c] per-tap slab
  __shared__ __bf16 Cis[32 * 128];  // [px][o] relu intermediate (bf16 A for stage2)
  __shared__ __bf16 B2s[32 * 128];  // [o][c] se_w2

  int bid = blockIdx.x;
  int x0 = (bid & 7) << 5;
  int y = (bid >> 3) & 255;
  int n = bid >> 11;
  int tid = threadIdx.x;
  int lane = tid & 31, wave = tid >> 5;
  int laneM = lane & 15, grp = lane >> 4;

  for (int i = tid; i < 64 * 102; i += 128) {
    int c = i / 102, row = i - c * 102;
    int r = row / 34, xx = row - r * 34;
    int gy = y + r - 1, gx = x0 + xx - 1;
    float v = 0.f;
    if ((unsigned)gy < 256u && (unsigned)gx < 256u)
      v = cf[(n * 64 + c) * HW + gy * IMG_W + gx];
    As[row * 64 + c] = (__bf16)v;
  }
  for (int i = tid; i < 32 * 128; i += 128) {
    int o = i >> 7, c = i & 127;
    B2s[i] = (__bf16)w2[o * 128 + c];
  }

  int o0 = wave * 32 + laneM;  // wave owns out channels [wave*32, wave*32+32)
  v8f acc00 = vsplat8(b1[o0]);
  v8f acc01 = vsplat8(b1[o0 + 16]);
  v8f acc10 = acc00, acc11 = acc01;

  for (int tap = 0; tap < 9; ++tap) {
    int ky = tap / 3, kx = tap - ky * 3;
    __syncthreads();
    for (int i = tid; i < 128 * 64; i += 128) {
      int o = i >> 6, c = i & 63;
      Bs[i] = (__bf16)w1[((o * 64 + c) * 3 + ky) * 3 + kx];
    }
    __syncthreads();
#pragma unroll
    for (int ks = 0; ks < 2; ++ks) {
      int c0 = ks * 32;
      const __bf16* ar = As + (ky * 34 + laneM + kx) * 64 + c0;
      v16bf a0 = frag_a(ar, grp);
      v16bf a1 = frag_a(ar + 16 * 64, grp);
      v16bf bA = frag_b(Bs + o0 * 64 + c0, grp);
      v16bf bB = frag_b(Bs + (o0 + 16) * 64 + c0, grp);
      acc00 = wmma_bf16(a0, bA, acc00);
      acc01 = wmma_bf16(a0, bB, acc01);
      acc10 = wmma_bf16(a1, bA, acc10);
      acc11 = wmma_bf16(a1, bB, acc11);
    }
  }
#pragma unroll
  for (int i = 0; i < 8; ++i) {
    int m0 = grp * 8 + i;
    Cis[m0 * 128 + o0]        = (__bf16)fmaxf(acc00[i], 0.f);
    Cis[m0 * 128 + o0 + 16]   = (__bf16)fmaxf(acc01[i], 0.f);
    Cis[(16 + m0) * 128 + o0]      = (__bf16)fmaxf(acc10[i], 0.f);
    Cis[(16 + m0) * 128 + o0 + 16] = (__bf16)fmaxf(acc11[i], 0.f);
  }
  __syncthreads();

  // stage 2: [32px x 32out] = [32px x 128] * [128 x 32]; one (mt,nt) per wave
  int mt = wave >> 1, nt = wave & 1;
  v8f acc2 = vsplat8(b2[nt * 16 + laneM]);
#pragma unroll
  for (int ks = 0; ks < 4; ++ks) {
    int c0 = ks * 32;
    v16bf a = frag_a(Cis + (mt * 16 + laneM) * 128 + c0, grp);
    v16bf b = frag_b(B2s + (nt * 16 + laneM) * 128 + c0, grp);
    acc2 = wmma_bf16(a, b, acc2);
  }
  int oc = nt * 16 + laneM;
  float* dst = out + (n * 32 + oc) * HW + y * IMG_W + x0 + mt * 16 + grp * 8;
#pragma unroll
  for (int i = 0; i < 8; ++i) dst[i] = acc2[i];
}

// ---------------------------------------------------------------------------
// 5) Poolings + tiny MLPs (exact f32)
// ---------------------------------------------------------------------------
__global__ void __launch_bounds__(256) k_pool_feat(const float* __restrict__ ef,
                                                   float* __restrict__ pooled) {
  __shared__ float red[256];
  int c = blockIdx.x;
  float s = 0.f;
  for (int i = threadIdx.x; i < HW; i += 256) s += ef[c * HW + i];
  red[threadIdx.x] = s;
  __syncthreads();
  for (int st = 128; st > 0; st >>= 1) {
    if (threadIdx.x < st) red[threadIdx.x] += red[threadIdx.x + st];
    __syncthreads();
  }
  if (threadIdx.x == 0) pooled[c] = red[0] * (1.f / 65536.f);
}

__global__ void __launch_bounds__(64) k_channel_mlp(
    const float* __restrict__ pooled, const float* __restrict__ w1,
    const float* __restrict__ w2, float* __restrict__ out) {
  __shared__ float hid[4];
  int t = threadIdx.x;
  if (t < 4) {
    float s = 0.f;
    for (int c = 0; c < 64; ++c) s += pooled[c] * w1[t * 64 + c];
    hid[t] = fmaxf(s, 0.f);
  }
  __syncthreads();
  float m = 0.f;
#pragma unroll
  for (int h = 0; h < 4; ++h) m += hid[h] * w2[t * 4 + h];
  out[t] = 1.f / (1.f + expf(-2.f * m));  // sigmoid(m + m)
}

__global__ void __launch_bounds__(256) k_pool_full(
    const float* __restrict__ cv, const float* __restrict__ cf,
    const float* __restrict__ cd, float* __restrict__ pooled) {
  __shared__ float red[256];
  int n = blockIdx.x / 80, c = blockIdx.x - n * 80;
  const float* src;
  if (c < 8)       src = cv + (n * 8 + c) * HW;
  else if (c < 72) src = cf + (n * 64 + (c - 8)) * HW;
  else             src = cd + (n * 8 + (c - 72)) * HW;
  float s = 0.f;
  for (int i = threadIdx.x; i < HW; i += 256) s += src[i];
  red[threadIdx.x] = s;
  __syncthreads();
  for (int st = 128; st > 0; st >>= 1) {
    if (threadIdx.x < st) red[threadIdx.x] += red[threadIdx.x + st];
    __syncthreads();
  }
  if (threadIdx.x == 0) pooled[blockIdx.x] = red[0] * (1.f / 65536.f);
}

__global__ void __launch_bounds__(32) k_gran_mlp(
    const float* __restrict__ pooled, const float* __restrict__ w1,
    const float* __restrict__ b1, const float* __restrict__ w2,
    const float* __restrict__ b2, float* __restrict__ out) {
  __shared__ float t32[32];
  int n = blockIdx.x, h = threadIdx.x;
  float s = b1[h];
  for (int c = 0; c < 80; ++c) s += pooled[n * 80 + c] * w1[h * 80 + c];
  t32[h] = fmaxf(s, 0.f);
  __syncthreads();
  if (h == 0) {
    float z[3], mx = -1e30f;
#pragma unroll
    for (int j = 0; j < 3; ++j) {
      float zz = b2[j];
      for (int hh = 0; hh < 32; ++hh) zz += t32[hh] * w2[j * 32 + hh];
      z[j] = zz;
      mx = fmaxf(mx, zz);
    }
    float e0 = expf(z[0] - mx), e1 = expf(z[1] - mx), e2 = expf(z[2] - mx);
    float inv = 1.f / (e0 + e1 + e2);
    out[n * 3 + 0] = e0 * inv;
    out[n * 3 + 1] = e1 * inv;
    out[n * 3 + 2] = e2 * inv;
  }
}

// ---------------------------------------------------------------------------
extern "C" void kernel_launch(void* const* d_in, const int* in_sizes, int n_in,
                              void* d_out, int out_size, void* d_ws, size_t ws_size,
                              hipStream_t stream) {
  (void)in_sizes; (void)n_in; (void)out_size; (void)ws_size;
  const float* ev   = (const float*)d_in[0];
  const float* ef   = (const float*)d_in[1];
  const float* ed   = (const float*)d_in[2];
  const float* cv   = (const float*)d_in[3];
  const float* cf   = (const float*)d_in[4];
  const float* cd   = (const float*)d_in[5];
  const float* qw   = (const float*)d_in[6];
  const float* chw1 = (const float*)d_in[7];
  const float* chw2 = (const float*)d_in[8];
  const float* kw1  = (const float*)d_in[9];
  const float* kw2  = (const float*)d_in[10];
  const float* kw3  = (const float*)d_in[11];
  const float* spw1 = (const float*)d_in[12];
  const float* spb1 = (const float*)d_in[13];
  const float* spw2 = (const float*)d_in[14];
  const float* spb2 = (const float*)d_in[15];
  const float* grw1 = (const float*)d_in[16];
  const float* grb1 = (const float*)d_in[17];
  const float* grw2 = (const float*)d_in[18];
  const float* grb2 = (const float*)d_in[19];
  const float* sew1 = (const float*)d_in[20];
  const float* seb1 = (const float*)d_in[21];
  const float* sew2 = (const float*)d_in[22];
  const float* seb2 = (const float*)d_in[23];

  float* out = (float*)d_out;
  float* o_contrib = out;                 // [5,3,256,256]
  float* o_qk  = out + 983040;            // [5,3,256,256]
  float* o_sp  = out + 1966080;           // [5,1,256,256]
  float* o_gr  = out + 2293760;           // [5,3,1,1]
  float* o_sem = out + 2293775;           // [5,32,256,256]
  float* o_ch  = out + 12779535;          // [1,64,1,1]

  float* ws = (float*)d_ws;
  float* pooled_feat = ws;                // 64 (fully rewritten each call)
  float* pooled_full = ws + 64;           // 400 (fully rewritten each call)

  k_contribution<<<(NAG * HW + 255) / 256, 256, 0, stream>>>(
      ev, ef, ed, cv, cf, cd, o_contrib);
  k_qk<<<NAG * 256 * 16, 32, 0, stream>>>(
      ef, cv, cf, cd, qw, kw1, kw2, kw3, o_qk);
  k_spatial<<<NAG * 256 * 8, 128, 0, stream>>>(
      cv, cf, cd, spw1, spb1, spw2, spb2, o_sp);
  k_semantic<<<NAG * 256 * 8, 128, 0, stream>>>(
      cf, sew1, seb1, sew2, seb2, o_sem);
  k_pool_feat<<<64, 256, 0, stream>>>(ef, pooled_feat);
  k_channel_mlp<<<1, 64, 0, stream>>>(pooled_feat, chw1, chw2, o_ch);
  k_pool_full<<<400, 256, 0, stream>>>(cv, cf, cd, pooled_full);
  k_gran_mlp<<<NAG, 32, 0, stream>>>(pooled_full, grw1, grb1, grw2, grb2, o_gr);
}
